// DualGATNetwork_26611617366628
// MI455X (gfx1250) — matrix operand order repaired and checked by reading, not compiled, not run on hardware
//
#include <hip/hip_runtime.h>
#include <hip/hip_bf16.h>
#include <float.h>

typedef __attribute__((ext_vector_type(16))) _Float16 v16h;
typedef __attribute__((ext_vector_type(8)))  float    v8f;

#define WMMA_F16(a, b, c) \
  __builtin_amdgcn_wmma_f32_16x16x32_f16(false, (a), false, (b), (short)0, (c), false, false)

// ---------------------------------------------------------------------------
// WMMA helpers (CDNA5 v_wmma_f32_16x16x32_f16, wave32)
//   A (16x32 f16): lane m=lane&15; lanes<16 hold K 0..7 & 16..23, lanes>=16
//   hold K 8..15 & 24..31 (ISA 7.12.2). A tile is row-major f16 in LDS.
//   B (32x16 f16): lane n=lane&15 holds K 0..7 & 16..23 (lanes>=16: K 8..15 & 24..31).
//   B comes from row-major [out][in] f16 weights (computing x @ W^T).
//   C/D: lane n=lane&15; VGPR j holds row m = j + (lane&16 ? 8 : 0).
// ---------------------------------------------------------------------------
__device__ __forceinline__ v16h ldsA_frag(const _Float16* tile, int stride, int lane, int k0) {
  const int m  = lane & 15;
  const int kb = k0 + ((lane & 16) ? 8 : 0);
  const _Float16* p = tile + m * stride + kb;
  v16h a;
#pragma unroll
  for (int i = 0; i < 8; ++i) { a[i] = p[i]; a[8 + i] = p[16 + i]; }
  return a;
}

__device__ __forceinline__ v16h glbB_frag(const _Float16* W, int kin, int lane, int n0, int k0) {
  const int n  = n0 + (lane & 15);
  const int kb = k0 + ((lane & 16) ? 8 : 0);
  const _Float16* p = W + (size_t)n * kin + kb;
  v16h b;
#pragma unroll
  for (int i = 0; i < 8; ++i) { b[i] = p[i]; b[8 + i] = p[16 + i]; }
  return b;
}

// dual-M-tile GEMM: one B fragment feeds two WMMAs (edges 0-15 and 16-31)
template <int NT, int KC>
__device__ __forceinline__ void gemm_acc2(v8f acc[2][NT], const _Float16* A, int astride,
                                          const _Float16* W, int kin, int n0, int lane) {
  for (int kc = 0; kc < KC; ++kc) {
    v16h a0 = ldsA_frag(A,                astride, lane, kc * 32);
    v16h a1 = ldsA_frag(A + 16 * astride, astride, lane, kc * 32);
#pragma unroll
    for (int j = 0; j < NT; ++j) {
      v16h b = glbB_frag(W, kin, lane, n0 + j * 16, kc * 32);
      acc[0][j] = WMMA_F16(a0, b, acc[0][j]);
      acc[1][j] = WMMA_F16(a1, b, acc[1][j]);
    }
  }
}

// single-M-tile variant (node kernel)
template <int NT, int KC>
__device__ __forceinline__ void gemm_acc(v8f* acc, const _Float16* A, int astride,
                                         const _Float16* W, int kin, int n0, int lane) {
  for (int kc = 0; kc < KC; ++kc) {
    v16h a = ldsA_frag(A, astride, lane, kc * 32);
#pragma unroll
    for (int j = 0; j < NT; ++j) {
      v16h b = glbB_frag(W, kin, lane, n0 + j * 16, kc * 32);
      acc[j] = WMMA_F16(a, b, acc[j]);
    }
  }
}

__device__ __forceinline__ void store_f16_tile(_Float16* dst, int stride, int lane, int n0,
                                               const v8f acc, const float* bias, bool relu) {
  const int n  = n0 + (lane & 15);
  const int mb = (lane & 16) ? 8 : 0;
  const float bv = bias[n];
#pragma unroll
  for (int j = 0; j < 8; ++j) {
    float v = acc[j] + bv;
    if (relu) v = fmaxf(v, 0.f);
    dst[(mb + j) * stride + n] = (_Float16)v;
  }
}

__device__ __forceinline__ void atomicMaxF(float* addr, float v) {
  int* ia = (int*)addr;
  int cur = __float_as_int(*addr);
  while (__int_as_float(cur) < v) {
    int prev = atomicCAS(ia, cur, __float_as_int(v));
    if (prev == cur) break;
    cur = prev;
  }
}

// CDNA5 async global->LDS copy (ASYNCcnt-tracked). Flat shared-pointer low
// 32 bits == LDS offset (ISA 10.2 aperture rules).
__device__ __forceinline__ void async_copy_b128(void* lds_dst, const void* gsrc) {
  unsigned l = (unsigned)(uintptr_t)lds_dst;
  asm volatile("global_load_async_to_lds_b128 %0, %1, off" :: "v"(l), "v"(gsrc) : "memory");
}
__device__ __forceinline__ void wait_async0() {
  asm volatile("s_wait_asynccnt 0" ::: "memory");
}

// ---------------------------------------------------------------------------
// Small utility kernels
// ---------------------------------------------------------------------------
__global__ void dgat_cvt_f16(_Float16* dst, const float* src, int n) {
  int i = blockIdx.x * blockDim.x + threadIdx.x;
  if (i < n) dst[i] = (_Float16)src[i];
}

__global__ void dgat_init_hash(unsigned* hkey, int* hval, int n) {
  int i = blockIdx.x * blockDim.x + threadIdx.x;
  if (i < n) { hkey[i] = 0xFFFFFFFFu; hval[i] = 0x7FFFFFFF; }
}

__global__ void dgat_init_node(float* agg, float* sum_out, float* sum_in,
                               float* cnt_out, float* cnt_in, int n128, int n) {
  int i = blockIdx.x * blockDim.x + threadIdx.x;
  if (i < n128) { agg[i] = -FLT_MAX; sum_out[i] = 0.f; sum_in[i] = 0.f; }
  if (i < n)    { cnt_out[i] = 0.f; cnt_in[i] = 0.f; }
}

__global__ void dgat_hash_insert(const int* row, const int* col, int E, int N,
                                 unsigned* hkey, int* hval, unsigned mask) {
  int e = blockIdx.x * blockDim.x + threadIdx.x;
  if (e >= E) return;
  unsigned key = (unsigned)row[e] * (unsigned)N + (unsigned)col[e];
  unsigned s = (key * 2654435761u) & mask;
  for (;;) {
    unsigned prev = atomicCAS(&hkey[s], 0xFFFFFFFFu, key);
    if (prev == 0xFFFFFFFFu || prev == key) { atomicMin(&hval[s], e); break; }
    s = (s + 1) & mask;
  }
}

__global__ void dgat_hash_lookup(const int* row, const int* col, int E, int N,
                                 const unsigned* hkey, const int* hval, unsigned mask,
                                 int* rev) {
  int e = blockIdx.x * blockDim.x + threadIdx.x;
  if (e >= E) return;
  unsigned key = (unsigned)col[e] * (unsigned)N + (unsigned)row[e];
  unsigned s = (key * 2654435761u) & mask;
  int r = -1;
  for (;;) {
    unsigned k = hkey[s];
    if (k == key) { r = hval[s]; break; }
    if (k == 0xFFFFFFFFu) break;
    s = (s + 1) & mask;
  }
  rev[e] = r;
}

__global__ void dgat_dmask(const float* pos, const int* row, const int* col,
                           const float* W1, const float* b1,
                           const float* W2, const float* b2,
                           float* dmask, int E) {
  int e = blockIdx.x * blockDim.x + threadIdx.x;
  if (e >= E) return;
  int r = row[e], c = col[e];
  float d0 = pos[r * 3 + 0] - pos[c * 3 + 0];
  float d1 = pos[r * 3 + 1] - pos[c * 3 + 1];
  float d2 = pos[r * 3 + 2] - pos[c * 3 + 2];
  float dist = sqrtf(d0 * d0 + d1 * d1 + d2 * d2);
  float f[4] = {d0, d1, d2, dist};
  float acc = b2[0];
  for (int o = 0; o < 32; ++o) {
    float s = b1[o];
#pragma unroll
    for (int k = 0; k < 4; ++k) s += W1[o * 4 + k] * f[k];
    acc += W2[o] * fmaxf(s, 0.f);
  }
  dmask[e] = 1.f / (1.f + __expf(-acc));
}

// ---------------------------------------------------------------------------
// Fused per-edge kernel: 32 edges per block, 8 waves (256 thr).
//   async ef tile -> LDS; gather x_i/rev/x_j; edge-update MLP (dual-M WMMA,
//   each B fragment feeds 2 WMMAs); q/k/v; per-(edge,head) attention;
//   prob out, scatter-max, segment sums, degree counts.
// LDS overlays (liveness): Cin(32K) -> Q|K|V|UE after qkv accumulation;
//   EFstage(16K f32) -> Hid(24K) before layer-1 stores.  Total < 64 KB.
// ---------------------------------------------------------------------------
__global__ __launch_bounds__(256) void dgat_edge(
    const float* __restrict__ x, const float* __restrict__ ef,
    const int* __restrict__ row, const int* __restrict__ col,
    const int* __restrict__ rev_idx, const float* __restrict__ dmask,
    const _Float16* euW1h, const float* euB1,
    const _Float16* euW2h, const float* euB2,
    const _Float16* pqWh, const float* pqB,
    const _Float16* pkWh, const float* pkB,
    const _Float16* pvWh, const float* pvB,
    const float* attW1, const float* attB1,
    const float* attW2, const float* attB2,
    float* outUE, float* outProb,
    float* agg, float* sum_out, float* sum_in,
    float* cnt_out, float* cnt_in, int E) {
  __shared__ __align__(16) char sBufA[32 * 512 * 2];  // Cin | {Q,K,V,UE}
  __shared__ __align__(16) char sBufB[32 * 384 * 2];  // EFstage(f32) | Hid
  __shared__ float sW1[32 * 32], sB1[32];
  __shared__ float sW2[16 * 32], sB2[16];
  __shared__ int   sRow[32], sCol[32], sRev[32];
  __shared__ float sDm[32];

  _Float16* sCin     = (_Float16*)sBufA;
  _Float16* sQ       = (_Float16*)sBufA;              //  8 KB
  _Float16* sK       = (_Float16*)(sBufA + 8192);     //  8 KB
  _Float16* sV       = (_Float16*)(sBufA + 16384);    //  8 KB
  _Float16* sUE      = (_Float16*)(sBufA + 24576);    //  8 KB
  float*    sEFstage = (float*)sBufB;                 // 16 KB
  _Float16* sHid     = (_Float16*)sBufB;              // 24 KB

  const int t    = threadIdx.x;
  const int lane = t & 31;
  const int wv   = t >> 5;
  const int e0   = blockIdx.x * 32;

  if (t < 32) {
    sRow[t] = row[e0 + t];
    sCol[t] = col[e0 + t];
    sRev[t] = rev_idx[e0 + t];
    sDm[t]  = dmask[e0 + t];
  }
  for (int i = t; i < 32 * 32; i += 256) sW1[i] = attW1[i];
  for (int i = t; i < 16 * 32; i += 256) sW2[i] = attW2[i];
  if (t < 32) sB1[t] = attB1[t];
  if (t < 16) sB2[t] = attB2[t];

  // async copy of the contiguous edge_feature tile (32x128 f32 = 16 KB)
#pragma unroll
  for (int r2 = 0; r2 < 4; ++r2) {
    int fo = r2 * 1024 + t * 4;
    async_copy_b128(&sEFstage[fo], ef + (size_t)e0 * 128 + fo);
  }
  __syncthreads();  // S0: sRow/sCol/sRev visible

  // gather x_i (cols 0-127) and x_j (cols 384-511)
  for (int idx = t; idx < 32 * 256; idx += 256) {
    int e = idx >> 8, c = idx & 255;
    float v = (c < 128) ? x[(size_t)sRow[e] * 128 + c]
                        : x[(size_t)sCol[e] * 128 + (c - 128)];
    int dstc = (c < 128) ? c : (256 + c);
    sCin[e * 512 + dstc] = (_Float16)v;
  }
  // gather reverse-edge features (cols 256-383)
  for (int idx = t; idx < 32 * 128; idx += 256) {
    int e = idx >> 7, c = idx & 127;
    int rr = sRev[e];
    float v = (rr >= 0) ? ef[(size_t)rr * 128 + c] : 0.f;
    sCin[e * 512 + 256 + c] = (_Float16)v;
  }
  wait_async0();
  __syncthreads();  // S1: all waves' async transfers complete
  // convert staged ef (f32) into Cin cols 128-255
  for (int idx = t; idx < 32 * 128; idx += 256) {
    sCin[(idx >> 7) * 512 + 128 + (idx & 127)] = (_Float16)sEFstage[idx];
  }
  __syncthreads();  // S2: Cin complete (EFstage now dead -> Hid may be written)

  // edge-update layer 1: [32,512] x [512,384] -> relu -> sHid
  {
    v8f acc[2][3] = {};
    gemm_acc2<3, 16>(acc, sCin, 512, euW1h, 512, wv * 48, lane);
#pragma unroll
    for (int mt = 0; mt < 2; ++mt)
#pragma unroll
      for (int j = 0; j < 3; ++j)
        store_f16_tile(sHid + mt * 16 * 384, 384, lane, wv * 48 + j * 16,
                       acc[mt][j], euB1, true);
  }

  // q/k/v projections: accumulate while Cin is live, store after barrier
  v8f pacc[3][2] = {};
  {
    const _Float16* aB[3] = {sCin, sCin + 128, sCin + 384};
    const _Float16* wB[3] = {pqWh, pkWh, pvWh};
#pragma unroll
    for (int m = 0; m < 3; ++m) {
      for (int kc = 0; kc < 4; ++kc) {
        v16h a0 = ldsA_frag(aB[m],            512, lane, kc * 32);
        v16h a1 = ldsA_frag(aB[m] + 16 * 512, 512, lane, kc * 32);
        v16h b  = glbB_frag(wB[m], 128, lane, wv * 16, kc * 32);
        pacc[m][0] = WMMA_F16(a0, b, pacc[m][0]);
        pacc[m][1] = WMMA_F16(a1, b, pacc[m][1]);
      }
    }
  }
  __syncthreads();  // S3: Cin dead everywhere; Hid complete

  // store q/k/v into the overlay
  {
    _Float16*    dB[3] = {sQ, sK, sV};
    const float* bB[3] = {pqB, pkB, pvB};
#pragma unroll
    for (int m = 0; m < 3; ++m)
#pragma unroll
      for (int mt = 0; mt < 2; ++mt)
        store_f16_tile(dB[m] + mt * 16 * 128, 128, lane, wv * 16, pacc[m][mt], bB[m], false);
  }

  // edge-update layer 2: [32,384] x [384,128] -> updated_edge (global + LDS)
  {
    v8f acc[2][1] = {};
    gemm_acc2<1, 12>(acc, sHid, 384, euW2h, 384, wv * 16, lane);
    int n  = wv * 16 + (lane & 15);
    int mb = (lane & 16) ? 8 : 0;
    float bv = euB2[n];
#pragma unroll
    for (int mt = 0; mt < 2; ++mt)
#pragma unroll
      for (int r = 0; r < 8; ++r) {
        float v = acc[mt][0][r] + bv;
        sUE[(mt * 16 + mb + r) * 128 + n] = (_Float16)v;
        outUE[(size_t)(e0 + mt * 16 + mb + r) * 128 + n] = v;
      }
  }
  __syncthreads();  // S4: Q/K/V/UE visible

  // attention: one thread per (edge, head)
  {
    const int e = t >> 3, h = t & 7;
    float a[32];
#pragma unroll
    for (int i = 0; i < 16; ++i) a[i]      = (float)sQ[e * 128 + i * 8 + h];
#pragma unroll
    for (int i = 0; i < 16; ++i) a[16 + i] = (float)sK[e * 128 + i * 8 + h];
    float h1[32];
    for (int o = 0; o < 32; ++o) {
      float s = sB1[o];
#pragma unroll 8
      for (int c = 0; c < 32; ++c) s += sW1[o * 32 + c] * a[c];
      h1[o] = fmaxf(s, 0.f);
    }
    const float dm = sDm[e];
    float att[16];
    float mx = -3.0e38f;
    for (int o = 0; o < 16; ++o) {
      float s = sB2[o];
#pragma unroll 8
      for (int c = 0; c < 32; ++c) s += sW2[o * 32 + c] * h1[c];
      s = s * dm * 0.25f;  // (att * mask) / TEMP, TEMP = 4
      att[o] = s;
      mx = fmaxf(mx, s);
    }
    float ssum = 0.f;
#pragma unroll
    for (int o = 0; o < 16; ++o) { float p = __expf(att[o] - mx); att[o] = p; ssum += p; }
    const float inv = 1.f / ssum;
    const int rg = sRow[e];
#pragma unroll
    for (int o = 0; o < 16; ++o) {
      float p = att[o] * inv;
      outProb[(size_t)(e0 + e) * 128 + o * 8 + h] = p;
      float w = p * (float)sV[e * 128 + o * 8 + h];
      atomicMaxF(&agg[(size_t)rg * 128 + o * 8 + h], w);
    }
  }

  // segment sums of updated_edge over row/col
  {
    const int c = t & 127, half = t >> 7;
#pragma unroll
    for (int e2 = 0; e2 < 16; ++e2) {
      int e = half * 16 + e2;
      float v = (float)sUE[e * 128 + c];
      atomicAdd(&sum_out[(size_t)sRow[e] * 128 + c], v);
      atomicAdd(&sum_in [(size_t)sCol[e] * 128 + c], v);
    }
  }
  if (t < 32) {
    atomicAdd(&cnt_out[sRow[t]], 1.f);
    atomicAdd(&cnt_in [sCol[t]], 1.f);
  }
}

// ---------------------------------------------------------------------------
// Fused per-node kernel: 16 nodes per block, 4 waves.
// ---------------------------------------------------------------------------
__global__ __launch_bounds__(128) void dgat_node(
    const float* __restrict__ x, const float* __restrict__ agg,
    const float* __restrict__ cnt_out, const float* __restrict__ cnt_in,
    const float* __restrict__ sum_out, const float* __restrict__ sum_in,
    const _Float16* nuW1h, const float* nuB1,
    const _Float16* nuW2h, const float* nuB2,
    const _Float16* eaWh,  const float* eaB,
    const _Float16* nlW1h, const float* nlB1,
    const _Float16* nlW2h, const float* nlB2,
    float* outFinal, int N) {
  __shared__ _Float16 sIn  [16 * 256];
  __shared__ _Float16 sHid [16 * 256];
  __shared__ _Float16 sUN  [16 * 128];
  __shared__ _Float16 sTwin[16 * 256];
  __shared__ float    sEA  [16 * 128];
  __shared__ _Float16 sNL  [16 * 128];

  const int t    = threadIdx.x;
  const int lane = t & 31;
  const int wv   = t >> 5;
  const int n0g  = blockIdx.x * 16;

  for (int idx = t; idx < 16 * 256; idx += 128) {
    int e = idx >> 8, c = idx & 255;
    int node = n0g + e;
    float co = cnt_out[node], ci = cnt_in[node];
    float v, tw;
    if (c < 128) {
      v  = x[(size_t)node * 128 + c];
      tw = sum_out[(size_t)node * 128 + c] / fmaxf(co, 1.f);
    } else {
      int cc = c - 128;
      v  = (co > 0.f) ? agg[(size_t)node * 128 + cc] : 0.f;
      tw = sum_in[(size_t)node * 128 + cc] / fmaxf(ci, 1.f);
    }
    sIn[idx]   = (_Float16)v;
    sTwin[idx] = (_Float16)tw;
  }
  __syncthreads();

  // nu layer 1: [16,256]x[256,256] -> relu
  {
    v8f acc[4] = {};
    gemm_acc<4, 8>(acc, sIn, 256, nuW1h, 256, wv * 64, lane);
#pragma unroll
    for (int j = 0; j < 4; ++j)
      store_f16_tile(sHid, 256, lane, wv * 64 + j * 16, acc[j], nuB1, true);
  }
  // edge attention: [16,256]x[256,128] -> sigmoid
  {
    v8f acc[2] = {};
    gemm_acc<2, 8>(acc, sTwin, 256, eaWh, 256, wv * 32, lane);
#pragma unroll
    for (int j = 0; j < 2; ++j) {
      int n  = wv * 32 + j * 16 + (lane & 15);
      int mb = (lane & 16) ? 8 : 0;
      float b = eaB[n];
#pragma unroll
      for (int r = 0; r < 8; ++r) {
        float v = acc[j][r] + b;
        sEA[(mb + r) * 128 + n] = 1.f / (1.f + __expf(-v));
      }
    }
  }
  __syncthreads();

  // nu layer 2: [16,256]x[256,128] (no relu)
  {
    v8f acc[2] = {};
    gemm_acc<2, 8>(acc, sHid, 256, nuW2h, 256, wv * 32, lane);
#pragma unroll
    for (int j = 0; j < 2; ++j)
      store_f16_tile(sUN, 128, lane, wv * 32 + j * 16, acc[j], nuB2, false);
  }
  __syncthreads();

  // nl layer 1: [16,128]x[128,128] -> relu
  {
    v8f acc[2] = {};
    gemm_acc<2, 4>(acc, sUN, 128, nlW1h, 128, wv * 32, lane);
#pragma unroll
    for (int j = 0; j < 2; ++j)
      store_f16_tile(sNL, 128, lane, wv * 32 + j * 16, acc[j], nlB1, true);
  }
  __syncthreads();

  // nl layer 2 * edge_attention -> final_node
  {
    v8f acc[2] = {};
    gemm_acc<2, 4>(acc, sNL, 128, nlW2h, 128, wv * 32, lane);
#pragma unroll
    for (int j = 0; j < 2; ++j) {
      int n  = wv * 32 + j * 16 + (lane & 15);
      int mb = (lane & 16) ? 8 : 0;
      float b = nlB2[n];
#pragma unroll
      for (int r = 0; r < 8; ++r) {
        float v = (acc[j][r] + b) * sEA[(mb + r) * 128 + n];
        outFinal[(size_t)(n0g + mb + r) * 128 + n] = v;
      }
    }
  }
}

// ---------------------------------------------------------------------------
// Launch
// ---------------------------------------------------------------------------
extern "C" void kernel_launch(void* const* d_in, const int* in_sizes, int n_in,
                              void* d_out, int out_size, void* d_ws, size_t ws_size,
                              hipStream_t stream) {
  const float* x    = (const float*)d_in[0];
  const float* ef   = (const float*)d_in[1];
  const float* pos  = (const float*)d_in[2];
  const int*   eix  = (const int*)d_in[3];
  const float* pqW  = (const float*)d_in[4];  const float* pqB  = (const float*)d_in[5];
  const float* pkW  = (const float*)d_in[6];  const float* pkB  = (const float*)d_in[7];
  const float* pvW  = (const float*)d_in[8];  const float* pvB  = (const float*)d_in[9];
  const float* dmW1 = (const float*)d_in[10]; const float* dmB1 = (const float*)d_in[11];
  const float* dmW2 = (const float*)d_in[12]; const float* dmB2 = (const float*)d_in[13];
  const float* euW1 = (const float*)d_in[14]; const float* euB1 = (const float*)d_in[15];
  const float* euW2 = (const float*)d_in[16]; const float* euB2 = (const float*)d_in[17];
  const float* atW1 = (const float*)d_in[18]; const float* atB1 = (const float*)d_in[19];
  const float* atW2 = (const float*)d_in[20]; const float* atB2 = (const float*)d_in[21];
  const float* nuW1 = (const float*)d_in[22]; const float* nuB1 = (const float*)d_in[23];
  const float* nuW2 = (const float*)d_in[24]; const float* nuB2 = (const float*)d_in[25];
  const float* eaW  = (const float*)d_in[26]; const float* eaB  = (const float*)d_in[27];
  const float* nlW1 = (const float*)d_in[28]; const float* nlB1 = (const float*)d_in[29];
  const float* nlW2 = (const float*)d_in[30]; const float* nlB2 = (const float*)d_in[31];

  const int N = in_sizes[0] / 128;
  const int E = in_sizes[3] / 2;
  const int* row = eix;
  const int* col = eix + E;

  float* outFinal = (float*)d_out;
  float* outUE    = outFinal + (size_t)N * 128;
  float* outProb  = outUE + (size_t)E * 128;

  // carve workspace
  char*  wsb = (char*)d_ws;
  size_t off = 0;
  auto carve = [&](size_t bytes) -> char* {
    char* p = wsb + off;
    off = (off + bytes + 255) & ~(size_t)255;
    return p;
  };
  _Float16* euW1h = (_Float16*)carve(sizeof(_Float16) * 384 * 512);
  _Float16* euW2h = (_Float16*)carve(sizeof(_Float16) * 128 * 384);
  _Float16* pqWh  = (_Float16*)carve(sizeof(_Float16) * 128 * 128);
  _Float16* pkWh  = (_Float16*)carve(sizeof(_Float16) * 128 * 128);
  _Float16* pvWh  = (_Float16*)carve(sizeof(_Float16) * 128 * 128);
  _Float16* nuW1h = (_Float16*)carve(sizeof(_Float16) * 256 * 256);
  _Float16* nuW2h = (_Float16*)carve(sizeof(_Float16) * 128 * 256);
  _Float16* eaWh  = (_Float16*)carve(sizeof(_Float16) * 128 * 256);
  _Float16* nlW1h = (_Float16*)carve(sizeof(_Float16) * 128 * 128);
  _Float16* nlW2h = (_Float16*)carve(sizeof(_Float16) * 128 * 128);

  const unsigned HSIZE = 1u << 20;
  const unsigned HMASK = HSIZE - 1;
  unsigned* hkey   = (unsigned*)carve(sizeof(unsigned) * HSIZE);
  int*      hval   = (int*)     carve(sizeof(int) * HSIZE);
  int*      rev    = (int*)     carve(sizeof(int) * (size_t)E);
  float*    dmask  = (float*)   carve(sizeof(float) * (size_t)E);
  float*    agg    = (float*)   carve(sizeof(float) * (size_t)N * 128);
  float*    sumOut = (float*)   carve(sizeof(float) * (size_t)N * 128);
  float*    sumIn  = (float*)   carve(sizeof(float) * (size_t)N * 128);
  float*    cntOut = (float*)   carve(sizeof(float) * (size_t)N);
  float*    cntIn  = (float*)   carve(sizeof(float) * (size_t)N);

  auto cvt = [&](_Float16* d, const float* s, int n) {
    dgat_cvt_f16<<<(n + 255) / 256, 256, 0, stream>>>(d, s, n);
  };
  cvt(euW1h, euW1, 384 * 512);
  cvt(euW2h, euW2, 128 * 384);
  cvt(pqWh,  pqW,  128 * 128);
  cvt(pkWh,  pkW,  128 * 128);
  cvt(pvWh,  pvW,  128 * 128);
  cvt(nuW1h, nuW1, 256 * 256);
  cvt(nuW2h, nuW2, 128 * 256);
  cvt(eaWh,  eaW,  128 * 256);
  cvt(nlW1h, nlW1, 128 * 128);
  cvt(nlW2h, nlW2, 128 * 128);

  dgat_init_hash<<<(HSIZE + 255) / 256, 256, 0, stream>>>(hkey, hval, (int)HSIZE);
  dgat_init_node<<<((N * 128) + 255) / 256, 256, 0, stream>>>(agg, sumOut, sumIn,
                                                              cntOut, cntIn, N * 128, N);
  dgat_hash_insert<<<(E + 255) / 256, 256, 0, stream>>>(row, col, E, N, hkey, hval, HMASK);
  dgat_hash_lookup<<<(E + 255) / 256, 256, 0, stream>>>(row, col, E, N, hkey, hval, HMASK, rev);
  dgat_dmask<<<(E + 255) / 256, 256, 0, stream>>>(pos, row, col, dmW1, dmB1, dmW2, dmB2,
                                                  dmask, E);

  dgat_edge<<<E / 32, 256, 0, stream>>>(
      x, ef, row, col, rev, dmask,
      euW1h, euB1, euW2h, euB2,
      pqWh, pqB, pkWh, pkB, pvWh, pvB,
      atW1, atB1, atW2, atB2,
      outUE, outProb,
      agg, sumOut, sumIn, cntOut, cntIn, E);

  dgat_node<<<N / 16, 128, 0, stream>>>(
      x, agg, cntOut, cntIn, sumOut, sumIn,
      nuW1h, nuB1, nuW2h, nuB2,
      eaWh, eaB, nlW1h, nlB1, nlW2h, nlB2,
      outFinal, N);

  (void)n_in; (void)out_size; (void)ws_size;
}
